// ChunkwiseRetention_6433861009417
// MI455X (gfx1250) — compile-verified, matrix operand order checked by reading, not run on hardware
//
#include <hip/hip_runtime.h>
#include <hip/hip_bf16.h>
#include <cmath>

#define Bb  4
#define Tt  2048
#define Hh  16
#define DHd 64
#define NEe 1024

typedef __bf16 v16bf __attribute__((ext_vector_type(16)));
typedef float  v8f   __attribute__((ext_vector_type(8)));
typedef unsigned int u32x4 __attribute__((ext_vector_type(4)));
typedef int i32x4 __attribute__((ext_vector_type(4)));
typedef int i32x8 __attribute__((ext_vector_type(8)));

union Frag16 {
  v16bf v;
  unsigned short u[16];
  uint4 q[2];
};

__device__ __forceinline__ unsigned short f2bf(float x) {
  union { float f; unsigned int i; } a; a.f = x;
  unsigned int r = a.i + 0x7FFFu + ((a.i >> 16) & 1u);
  return (unsigned short)(r >> 16);
}

// A fragment 16x32 (bf16) from row-major src [rows][ld], tile at (row0, k0).
// ISA 7.12.2: lane L holds row L&15, K-groups {kb..kb+7, kb+16..kb+23}, kb = 8*(L>>4).
__device__ __forceinline__ void load_a16(Frag16& f, const unsigned short* src, int ld, int row0, int k0) {
  int lane = threadIdx.x & 31;
  const unsigned short* p = src + (size_t)(row0 + (lane & 15)) * ld + k0 + ((lane >> 4) << 3);
  f.q[0] = *(const uint4*)p;
  f.q[1] = *(const uint4*)(p + 16);
}

// B fragment 32x16 from K-transposed storage srcT[n][k] (contiguous per lane).
__device__ __forceinline__ void load_bT16(Frag16& f, const unsigned short* srcT, int ld, int n0, int k0) {
  int lane = threadIdx.x & 31;
  const unsigned short* p = srcT + (size_t)(n0 + (lane & 15)) * ld + k0 + ((lane >> 4) << 3);
  f.q[0] = *(const uint4*)p;
  f.q[1] = *(const uint4*)(p + 16);
}

__device__ __forceinline__ v8f wmma_bf16(const Frag16& a, const Frag16& b, v8f c) {
  return __builtin_amdgcn_wmma_f32_16x16x32_bf16(false, a.v, false, b.v, (short)0, c, false, false);
}

// TDM: 2-D tensor tile load into LDS (D# per ISA cdna5 §8.3/8.4; bf16 elements).
__device__ __forceinline__ void tdm_load_2d(unsigned lds_off, const void* gptr,
                                            unsigned tensor_d0, unsigned tensor_d1,
                                            unsigned tile_d0, unsigned tile_d1,
                                            unsigned stride0) {
  unsigned long long ga = (unsigned long long)(uintptr_t)gptr;
  u32x4 g0;
  g0[0] = 1u;                                            // count=1, user descriptor
  g0[1] = lds_off;                                       // lds_addr (bytes)
  g0[2] = (unsigned)(ga & 0xffffffffull);                // global_addr[31:0] (bits 95:64)
  g0[3] = (unsigned)((ga >> 32) & 0x01ffffffull)         // global_addr[56:32]
        | (2u << 30);                                    // type = 2 ("image")
  i32x8 g1;
  g1[0] = (int)(1u << 16);                               // wg_mask=0, data_size=1 (2 bytes)
  g1[1] = (int)(tensor_d0 << 16);                        // tensor_dim0[15:0] at bit 48
  g1[2] = (int)((tensor_d0 >> 16) | (tensor_d1 << 16));  // dim0 hi | dim1 lo
  g1[3] = (int)((tensor_d1 >> 16) | (tile_d0 << 16));    // dim1 hi | tile_dim0
  g1[4] = (int)(tile_d1 & 0xffffu);                      // tile_dim1, tile_dim2=0
  g1[5] = (int)stride0;                                  // tensor_dim0_stride[31:0]
  g1[6] = 0;                                             // stride0 hi | dim1_stride lo
  g1[7] = 0;
  i32x4 z4 = (i32x4)0;
#if defined(__clang_major__) && (__clang_major__ >= 23)
  i32x8 z8 = (i32x8)0;
  __builtin_amdgcn_tensor_load_to_lds(g0, g1, z4, z4, z8, 0);
#else
  __builtin_amdgcn_tensor_load_to_lds(g0, g1, z4, z4, 0);
#endif
}

// ---------------- casts ----------------
__global__ void cr_cast_x(const float* __restrict__ x, unsigned short* __restrict__ xbf, float* sums) {
  size_t i = (size_t)blockIdx.x * 256 + threadIdx.x;
  if (i < (size_t)Bb * Tt * NEe) xbf[i] = f2bf(x[i]);
  if (blockIdx.x == 0 && threadIdx.x < 16) sums[threadIdx.x] = 0.f;  // zero GN scratch
}

__global__ void cr_cast_w(const float* __restrict__ Wq, const float* __restrict__ Wk,
                          const float* __restrict__ Wv, unsigned short* __restrict__ wt) {
  int z = blockIdx.z;
  const float* W = (z == 0) ? Wq : ((z == 1) ? Wk : Wv);
  size_t i = (size_t)blockIdx.x * 256 + threadIdx.x;     // over NE*NE
  int k = (int)(i >> 10), n = (int)(i & 1023);
  wt[(size_t)z * NEe * NEe + (size_t)n * NEe + k] = f2bf(W[i]);   // store transposed
}

__global__ void cr_cast_pkv(const float* __restrict__ pkv, unsigned short* __restrict__ pkvT) {
  int i = blockIdx.x * 256 + threadIdx.x;                // H*64*64
  int h = i >> 12, r = i & 4095, d = r >> 6, e = r & 63;
  pkvT[(h << 12) + (e << 6) + d] = f2bf(pkv[i]);         // store transposed
}

// ---------------- projections: Q/K/V = X @ W ----------------
// z==0 -> Q row-major [b,h,t,d]                 (A-frags of QK^T, pp)
// z==1 -> K row-major [b,h,t,d] + K^T [b,h,d,t] (B-frags of QK^T; A-frags of K^T V)
// z==2 -> V^T [b,h,d,t] only                    (B-frags of S*V and K^T V)
__global__ void __launch_bounds__(256) cr_proj(const unsigned short* __restrict__ xbf,
                                               const unsigned short* __restrict__ wt,
                                               unsigned short* __restrict__ qbf,
                                               unsigned short* __restrict__ kbf,
                                               unsigned short* __restrict__ ktbf,
                                               unsigned short* __restrict__ vtbf) {
  int wave = threadIdx.x >> 5, lane = threadIdx.x & 31;
  int row0 = (blockIdx.y * 8 + wave) * 16;
  int col0 = blockIdx.x * 64;
  int z = blockIdx.z;
  const unsigned short* W = wt + (size_t)z * NEe * NEe;

  v8f acc[4];
#pragma unroll
  for (int c = 0; c < 4; ++c)
#pragma unroll
    for (int r = 0; r < 8; ++r) acc[c][r] = 0.f;

  for (int k0 = 0; k0 < NEe; k0 += 32) {
    Frag16 a; load_a16(a, xbf, NEe, row0, k0);
#pragma unroll
    for (int ct = 0; ct < 4; ++ct) {
      Frag16 bw; load_bT16(bw, W, NEe, col0 + ct * 16, k0);
      acc[ct] = wmma_bf16(a, bw, acc[ct]);
    }
  }
  int nn = lane & 15, mhi = (lane >> 4) << 3;
#pragma unroll
  for (int ct = 0; ct < 4; ++ct)
#pragma unroll
    for (int r = 0; r < 8; ++r) {
      int grow = row0 + r + mhi;
      int col  = col0 + ct * 16 + nn;
      int b = grow >> 11, t = grow & 2047;
      int h = col >> 6,  d = col & 63;
      size_t bh = (size_t)(b * Hh + h);
      unsigned short val = f2bf(acc[ct][r]);
      if (z == 0) {
        qbf[(bh * Tt + t) * DHd + d] = val;
      } else if (z == 1) {
        kbf [(bh * Tt + t) * DHd + d] = val;
        ktbf[(bh * DHd + d) * Tt + t] = val;
      } else {
        vtbf[(bh * DHd + d) * Tt + t] = val;
      }
    }
}

// ---------------- pp^T = (Q @ past_kv)^T : stored [b,h,e,t] ----------------
__global__ void __launch_bounds__(256) cr_pp(const unsigned short* __restrict__ qbf,
                                             const unsigned short* __restrict__ pkvT,
                                             unsigned short* __restrict__ pptbf) {
  int wave = threadIdx.x >> 5, lane = threadIdx.x & 31;
  int wid = blockIdx.x * 8 + wave;
  int bh = wid >> 7, n0 = (wid & 127) << 4, h = bh & 15;
  const unsigned short* Q  = qbf + (size_t)bh * Tt * DHd;
  const unsigned short* PK = pkvT + (h << 12);

  v8f acc[4];
#pragma unroll
  for (int c = 0; c < 4; ++c)
#pragma unroll
    for (int r = 0; r < 8; ++r) acc[c][r] = 0.f;

#pragma unroll
  for (int ks = 0; ks < 2; ++ks) {
    Frag16 a; load_a16(a, Q, DHd, n0, ks * 32);
#pragma unroll
    for (int ct = 0; ct < 4; ++ct) {
      Frag16 bb; load_bT16(bb, PK, DHd, ct * 16, ks * 32);
      acc[ct] = wmma_bf16(a, bb, acc[ct]);
    }
  }
  int nn = lane & 15, mhi = (lane >> 4) << 3;
#pragma unroll
  for (int ct = 0; ct < 4; ++ct)
#pragma unroll
    for (int r = 0; r < 8; ++r) {
      int t = n0 + r + mhi;
      int e = ct * 16 + nn;
      pptbf[((size_t)bh * DHd + e) * Tt + t] = f2bf(acc[ct][r]);
    }
}

// ---------------- retention: inner (QK^T * D) V  +  cross D^T pp ----------------
__global__ void __launch_bounds__(256) cr_retention(const unsigned short* __restrict__ qbf,
                                                    const unsigned short* __restrict__ kbf,
                                                    const unsigned short* __restrict__ vtbf,
                                                    const unsigned short* __restrict__ pptbf,
                                                    float* __restrict__ ret) {
  __shared__ __align__(16) unsigned short stile[8][16 * 32];
  int wave = threadIdx.x >> 5, lane = threadIdx.x & 31;
  int wid = blockIdx.x * 8 + wave;
  int bh = wid >> 7, n0 = (wid & 127) << 4, h = bh & 15;
  float log2g = log2f(1.f - exp2f(-5.f - (float)h));
  size_t base  = (size_t)bh * Tt * DHd;
  const unsigned short* Q  = qbf  + base;
  const unsigned short* K  = kbf  + base;
  const unsigned short* VT = vtbf + base;   // [d][t]
  const unsigned short* PT = pptbf + base;  // [e][t]

  Frag16 aq0, aq1;
  load_a16(aq0, Q, DHd, n0, 0);
  load_a16(aq1, Q, DHd, n0, 32);

  v8f acc[4];
#pragma unroll
  for (int c = 0; c < 4; ++c)
#pragma unroll
    for (int r = 0; r < 8; ++r) acc[c][r] = 0.f;

  int nn = lane & 15, mhi = (lane >> 4) << 3, kb = (lane >> 4) << 3;
  unsigned short* my = &stile[wave][0];

  // ---- intra-chunk term: causal, 32 s-columns per iteration ----
  for (int s0 = 0; s0 <= n0; s0 += 32) {
    __builtin_prefetch(K + (size_t)(s0 + 32) * DHd, 0, 1);
#pragma unroll
    for (int ss = 0; ss < 2; ++ss) {
      v8f sf;
#pragma unroll
      for (int r = 0; r < 8; ++r) sf[r] = 0.f;
      Frag16 bk;
      load_bT16(bk, K, DHd, s0 + ss * 16, 0);  sf = wmma_bf16(aq0, bk, sf);
      load_bT16(bk, K, DHd, s0 + ss * 16, 32); sf = wmma_bf16(aq1, bk, sf);
      // apply decay D[t,s] = gamma^(t-s) (t>=s) in C layout, spill to LDS as [m][32]
#pragma unroll
      for (int r = 0; r < 8; ++r) {
        int t = n0 + r + mhi;
        int s = s0 + ss * 16 + nn;
        float dcy = (t >= s) ? exp2f((float)(t - s) * log2g) : 0.f;
        my[(r + mhi) * 32 + ss * 16 + nn] = f2bf(sf[r] * dcy);
      }
    }
    // LDS is in-order per wave; fence the compiler + drain DScnt before re-reading
    asm volatile("s_wait_dscnt 0" ::: "memory");
    Frag16 sa;
    sa.q[0] = *(const uint4*)&my[nn * 32 + kb];
    sa.q[1] = *(const uint4*)&my[nn * 32 + kb + 16];
#pragma unroll
    for (int ct = 0; ct < 4; ++ct) {
      Frag16 bv; load_bT16(bv, VT, Tt, ct * 16, s0);   // B(s,d) = V^T[d][s], contiguous
      acc[ct] = wmma_bf16(sa, bv, acc[ct]);
    }
  }

  // ---- cross term: cross[t] = sum_{s>=t} gamma^(s-t) * pp[s]; decay A-frag in regs ----
  int tt = n0 + nn;
  for (int s0 = n0 & ~31; s0 < Tt; s0 += 32) {
    __builtin_prefetch(PT + s0 + 32, 0, 1);
    Frag16 da;
#pragma unroll
    for (int j = 0; j < 8; ++j) {
      int s1 = s0 + kb + j;
      int s2 = s0 + kb + 16 + j;
      da.u[j]     = (s1 >= tt) ? f2bf(exp2f((float)(s1 - tt) * log2g)) : (unsigned short)0;
      da.u[8 + j] = (s2 >= tt) ? f2bf(exp2f((float)(s2 - tt) * log2g)) : (unsigned short)0;
    }
#pragma unroll
    for (int ct = 0; ct < 4; ++ct) {
      Frag16 bp; load_bT16(bp, PT, Tt, ct * 16, s0);   // B(s,e) = pp^T[e][s], contiguous
      acc[ct] = wmma_bf16(da, bp, acc[ct]);
    }
  }

#pragma unroll
  for (int ct = 0; ct < 4; ++ct)
#pragma unroll
    for (int r = 0; r < 8; ++r)
      ret[base + (size_t)(n0 + r + mhi) * DHd + ct * 16 + nn] = acc[ct][r];
}

// ---------------- state update: gamma*past_kv + mean_b K^T V (TDM-staged) ----------------
// One workgroup per head h. Wave 0 DMAs 64x256 bf16 panels of K^T and V^T into LDS with
// the Tensor Data Mover; all 8 waves then run WMMA out of LDS.
__global__ void __launch_bounds__(256) cr_kv(const unsigned short* __restrict__ ktbf,
                                             const unsigned short* __restrict__ vtbf,
                                             const float* __restrict__ past_kv,
                                             float* __restrict__ out_kv) {
  __shared__ __align__(16) unsigned short lds_k[64 * 256];   // 32 KB
  __shared__ __align__(16) unsigned short lds_v[64 * 256];   // 32 KB
  int wave = threadIdx.x >> 5, lane = threadIdx.x & 31;
  int h  = blockIdx.x;
  int m0 = (wave & 3) * 16;            // d-subtile
  int e0 = (wave >> 2) * 32;           // e-half

  v8f acc[2];
#pragma unroll
  for (int c = 0; c < 2; ++c)
#pragma unroll
    for (int r = 0; r < 8; ++r) acc[c][r] = 0.f;

  unsigned ldsk_off = (unsigned)(uintptr_t)(void*)&lds_k[0];
  unsigned ldsv_off = (unsigned)(uintptr_t)(void*)&lds_v[0];

  for (int b = 0; b < Bb; ++b) {
    size_t panel = (size_t)(b * Hh + h) * DHd * Tt;
    for (int tc = 0; tc < Tt; tc += 256) {
      __syncthreads();                 // previous chunk fully consumed
      if (wave == 0) {
        tdm_load_2d(ldsk_off, ktbf + panel + tc, Tt, DHd, 256, DHd, Tt);
        tdm_load_2d(ldsv_off, vtbf + panel + tc, Tt, DHd, 256, DHd, Tt);
        __builtin_amdgcn_s_wait_tensorcnt((short)0);
      }
      __syncthreads();                 // data visible to all waves
      for (int t0 = 0; t0 < 256; t0 += 32) {
        Frag16 a;
        { const unsigned short* p = &lds_k[(m0 + (lane & 15)) * 256 + t0 + ((lane >> 4) << 3)];
          a.q[0] = *(const uint4*)p; a.q[1] = *(const uint4*)(p + 16); }
#pragma unroll
        for (int ct = 0; ct < 2; ++ct) {
          Frag16 bv;
          { const unsigned short* p = &lds_v[(e0 + ct * 16 + (lane & 15)) * 256 + t0 + ((lane >> 4) << 3)];
            bv.q[0] = *(const uint4*)p; bv.q[1] = *(const uint4*)(p + 16); }
          acc[ct] = wmma_bf16(a, bv, acc[ct]);
        }
      }
    }
  }
  float gamma = 1.f - exp2f(-5.f - (float)h);
  int nn = lane & 15, mhi = (lane >> 4) << 3;
#pragma unroll
  for (int ct = 0; ct < 2; ++ct)
#pragma unroll
    for (int r = 0; r < 8; ++r) {
      int d = m0 + r + mhi;
      int e = e0 + ct * 16 + nn;
      size_t o = ((size_t)h * DHd + d) * DHd + e;
      out_kv[o] = gamma * past_kv[o] + 0.25f * acc[ct][r];
    }
}

// ---------------- GroupNorm(1,H): two-pass over ret ----------------
__global__ void cr_gn_partial(const float* __restrict__ ret, float* __restrict__ sums) {
  __shared__ float rs[256], rq[256];
  int b = blockIdx.y;
  const int N = Hh * Tt * DHd;
  const float* p = ret + (size_t)b * N;
  float s = 0.f, s2 = 0.f;
  for (int i = blockIdx.x * blockDim.x + threadIdx.x; i < N; i += blockDim.x * gridDim.x) {
    float v = p[i]; s += v; s2 += v * v;
  }
  rs[threadIdx.x] = s; rq[threadIdx.x] = s2;
  __syncthreads();
  for (int off = 128; off > 0; off >>= 1) {
    if (threadIdx.x < off) { rs[threadIdx.x] += rs[threadIdx.x + off]; rq[threadIdx.x] += rq[threadIdx.x + off]; }
    __syncthreads();
  }
  if (threadIdx.x == 0) {
    atomicAdd(&sums[b * 2 + 0], rs[0]);
    atomicAdd(&sums[b * 2 + 1], rq[0]);
  }
}

__global__ void cr_gn_out(const float* __restrict__ ret, const float* __restrict__ sums,
                          const float* __restrict__ gw, const float* __restrict__ gb,
                          float* __restrict__ out) {
  size_t idx = (size_t)blockIdx.x * 256 + threadIdx.x;     // over B*T*NE
  int b = (int)(idx >> 21);
  int rem = (int)(idx & ((1u << 21) - 1));
  int t = rem >> 10;
  int c = rem & 1023;
  int h = c >> 6, d = c & 63;
  const float invn = 1.f / (float)(Hh * Tt * DHd);
  float mu  = sums[b * 2 + 0] * invn;
  float var = sums[b * 2 + 1] * invn - mu * mu;
  float v = ret[(((size_t)(b * Hh + h)) * Tt + t) * DHd + d];
  out[idx] = (v - mu) * rsqrtf(var + 1e-5f) * gw[h] + gb[h];
}

extern "C" void kernel_launch(void* const* d_in, const int* in_sizes, int n_in,
                              void* d_out, int out_size, void* d_ws, size_t ws_size,
                              hipStream_t stream) {
  const float* x   = (const float*)d_in[0];
  const float* pkv = (const float*)d_in[1];
  const float* Wq  = (const float*)d_in[2];
  const float* Wk  = (const float*)d_in[3];
  const float* Wv  = (const float*)d_in[4];
  const float* gw  = (const float*)d_in[5];
  const float* gb  = (const float*)d_in[6];
  float* out    = (float*)d_out;
  float* out_kv = out + (size_t)Bb * Tt * NEe;

  const size_t BHTD = (size_t)Bb * Hh * Tt * DHd;   // 8,388,608
  char* ws = (char*)d_ws;
  size_t off = 0;
  unsigned short* XBF  = (unsigned short*)(ws + off); off += (size_t)Bb * Tt * NEe * 2;
  unsigned short* WT   = (unsigned short*)(ws + off); off += (size_t)3 * NEe * NEe * 2;
  unsigned short* PKVT = (unsigned short*)(ws + off); off += (size_t)Hh * 64 * 64 * 2;
  unsigned short* QBF  = (unsigned short*)(ws + off); off += BHTD * 2;
  unsigned short* KBF  = (unsigned short*)(ws + off); off += BHTD * 2;
  unsigned short* KTBF = (unsigned short*)(ws + off); off += BHTD * 2;
  unsigned short* VTBF = (unsigned short*)(ws + off); off += BHTD * 2;
  unsigned short* PPT  = (unsigned short*)(ws + off); off += BHTD * 2;
  float*          RET  = (float*)(ws + off);          off += BHTD * 4;
  float*          SUMS = (float*)(ws + off);          off += 64 * 4;

  cr_cast_x    <<<32768, 256, 0, stream>>>(x, XBF, SUMS);
  cr_cast_w    <<<dim3(4096, 1, 3), 256, 0, stream>>>(Wq, Wk, Wv, WT);
  cr_cast_pkv  <<<256, 256, 0, stream>>>(pkv, PKVT);
  cr_proj      <<<dim3(16, 64, 3), 256, 0, stream>>>(XBF, WT, QBF, KBF, KTBF, VTBF);
  cr_pp        <<<1024, 256, 0, stream>>>(QBF, PKVT, PPT);
  cr_retention <<<1024, 256, 0, stream>>>(QBF, KBF, VTBF, PPT, RET);
  cr_kv        <<<16, 256, 0, stream>>>(KTBF, VTBF, pkv, out_kv);
  cr_gn_partial<<<dim3(256, 4), 256, 0, stream>>>(RET, SUMS);
  cr_gn_out    <<<32768, 256, 0, stream>>>(RET, SUMS, gw, gb, out);
}
